// BilinearGrounding_21792664060129
// MI455X (gfx1250) — matrix operand order repaired and checked by reading, not compiled, not run on hardware
//
#include <hip/hip_runtime.h>

typedef __attribute__((ext_vector_type(16))) __bf16 v16bf;
typedef __attribute__((ext_vector_type(8)))  __bf16 v8bf;
typedef __attribute__((ext_vector_type(8)))  float  v8f;
typedef __attribute__((ext_vector_type(4)))  float  v4f;

#define BM 128
#define BN 128
#define BK 32
#define LDSS 40   // LDS row stride in bf16 (80B: keeps all 16B chunks aligned, staggers banks)

struct BF16x16 { v8bf lo, hi; };
struct RawF32x16 { v4f a, b, c, d; };

// ---- raw (unconverted) global tile loads: issue early, convert late ----
__device__ __forceinline__ RawF32x16 raw_load16(const float* p, bool valid) {
  RawF32x16 r;
  if (valid) {
    const v4f* q = (const v4f*)p;  // 64B-aligned by construction (offsets are x16 floats)
    r.a = q[0]; r.b = q[1]; r.c = q[2]; r.d = q[3];
  } else {
    const v4f z = {0.f, 0.f, 0.f, 0.f};
    r.a = z; r.b = z; r.c = z; r.d = z;
  }
  return r;
}

__device__ __forceinline__ BF16x16 raw_load16(const __bf16* p, bool valid) {
  BF16x16 r;
  if (valid) {
    const v8bf* q = (const v8bf*)p;  // 32B-aligned by construction
    r.lo = q[0]; r.hi = q[1];
  } else {
#pragma unroll
    for (int j = 0; j < 8; ++j) { r.lo[j] = (__bf16)0.0f; r.hi[j] = (__bf16)0.0f; }
  }
  return r;
}

__device__ __forceinline__ BF16x16 to_bf(const RawF32x16& f) {
  BF16x16 r;
#pragma unroll
  for (int j = 0; j < 4; ++j) {
    r.lo[j]     = (__bf16)f.a[j];
    r.lo[4 + j] = (__bf16)f.b[j];
    r.hi[j]     = (__bf16)f.c[j];
    r.hi[4 + j] = (__bf16)f.d[j];
  }
  return r;
}
__device__ __forceinline__ BF16x16 to_bf(const BF16x16& f) { return f; }

// TN GEMM macro-tile: C[m,n] = sum_k A[m,k] * B[n,k], both K-contiguous.
// 256 threads / 8 waves; wave grid 4(M) x 2(N); per-wave 32x64 = 2x4 WMMA accs.
// Software-pipelined: chunk i+1 global loads issued before chunk i's WMMAs;
// double-buffered LDS -> one barrier per K-step.
template <typename TA, typename TB>
__device__ __forceinline__ void gemm_tn_tile(
    const TA* __restrict__ A, int lda,
    const TB* __restrict__ B, int ldb,
    int M, int N, int K, int blockM, int blockN, v8f acc[2][4])
{
  __shared__ __bf16 sA[2][BM][LDSS];
  __shared__ __bf16 sB[2][BN][LDSS];

  const int tid  = threadIdx.x;
  const int lane = tid & 31;
  const int w    = tid >> 5;
  const int wm   = w & 3;        // 0..3  -> M offset wm*32
  const int wn   = w >> 2;       // 0..1  -> N offset wn*64
  const int half = lane >> 4;    // 0/1
  const int lrow = lane & 15;
  const int ldRow = tid >> 1;          // 0..127 : tile row
  const int ldCol = (tid & 1) << 4;    // 0 or 16 : K sub-chunk

  const v8f vzero = {0.f,0.f,0.f,0.f,0.f,0.f,0.f,0.f};
#pragma unroll
  for (int mi = 0; mi < 2; ++mi)
#pragma unroll
    for (int ni = 0; ni < 4; ++ni) acc[mi][ni] = vzero;

  const int  gm = blockM + ldRow;
  const int  gn = blockN + ldRow;
  const bool va = gm < M;
  const bool vb = gn < N;
  const TA* pA = A + (size_t)gm * lda + ldCol;
  const TB* pB = B + (size_t)gn * ldb + ldCol;

  const int nk = K / BK;   // K is a multiple of 32 in all three GEMMs

  // ---- prologue: stage chunk 0 ----
  auto ra = raw_load16(pA, va);
  auto rb = raw_load16(pB, vb);
  {
    BF16x16 t = to_bf(ra);
    *(v8bf*)&sA[0][ldRow][ldCol + 0] = t.lo;
    *(v8bf*)&sA[0][ldRow][ldCol + 8] = t.hi;
    t = to_bf(rb);
    *(v8bf*)&sB[0][ldRow][ldCol + 0] = t.lo;
    *(v8bf*)&sB[0][ldRow][ldCol + 8] = t.hi;
  }
  __syncthreads();

  for (int i = 0; i < nk; ++i) {
    const int  cur = i & 1;
    const bool hn  = (i + 1) < nk;

    // Issue next chunk's global loads NOW; s_wait_loadcnt lands after the WMMAs.
    if (hn) {
      ra = raw_load16(pA + (size_t)(i + 1) * BK, va);
      rb = raw_load16(pB + (size_t)(i + 1) * BK, vb);
    }

    // ---- compute chunk i from LDS ----
    union FragU { v16bf v; v8bf h[2]; };
    v16bf afr[2], bfr[4];
#pragma unroll
    for (int mi = 0; mi < 2; ++mi) {
      // A 16x32 bf16 layout: lanes 0-15 hold K{0-7,16-23}, lanes 16-31 K{8-15,24-31}
      const __bf16* pa = &sA[cur][wm * 32 + mi * 16 + lrow][half * 8];
      FragU u; u.h[0] = *(const v8bf*)pa; u.h[1] = *(const v8bf*)(pa + 16);
      afr[mi] = u.v;
    }
#pragma unroll
    for (int ni = 0; ni < 4; ++ni) {
      // B 32x16 bf16 layout: lane = column; lanes 0-15 K{0-7,16-23}, lanes 16-31 K{8-15,24-31}
      const __bf16* pb = &sB[cur][wn * 64 + ni * 16 + lrow][half * 8];
      FragU u; u.h[0] = *(const v8bf*)pb; u.h[1] = *(const v8bf*)(pb + 16);
      bfr[ni] = u.v;
    }
#pragma unroll
    for (int mi = 0; mi < 2; ++mi)
#pragma unroll
      for (int ni = 0; ni < 4; ++ni)
        acc[mi][ni] = __builtin_amdgcn_wmma_f32_16x16x32_bf16(
            false, afr[mi], false, bfr[ni], (short)0, acc[mi][ni], false, false);

    // ---- stage chunk i+1 into the other buffer ----
    if (hn) {
      const int nxt = cur ^ 1;
      BF16x16 t = to_bf(ra);
      *(v8bf*)&sA[nxt][ldRow][ldCol + 0] = t.lo;
      *(v8bf*)&sA[nxt][ldRow][ldCol + 8] = t.hi;
      t = to_bf(rb);
      *(v8bf*)&sB[nxt][ldRow][ldCol + 0] = t.lo;
      *(v8bf*)&sB[nxt][ldRow][ldCol + 8] = t.hi;
    }
    __syncthreads();
  }
}

// Kernel 1: encI_p[m, t] = encI[m,:2048] . K_w[t,:2048] + K_b[t]   (m = b*100+r)
__global__ void __launch_bounds__(256)
k_proj(const float* __restrict__ encI, const float* __restrict__ Kw,
       const float* __restrict__ Kb, __bf16* __restrict__ out)
{
  v8f acc[2][4];
  const int blockN = blockIdx.x * BN, blockM = blockIdx.y * BM;
  gemm_tn_tile<float, float>(encI, 2048, Kw, 2048, 6400, 768, 2048, blockM, blockN, acc);

  const int lane = threadIdx.x & 31, w = threadIdx.x >> 5;
  const int wm = w & 3, wn = w >> 2, half = lane >> 4, lrow = lane & 15;
#pragma unroll
  for (int mi = 0; mi < 2; ++mi)
#pragma unroll
    for (int ni = 0; ni < 4; ++ni) {
      const int n = blockN + wn * 64 + ni * 16 + lrow;
      const float kb = Kb[n];
#pragma unroll
      for (int v = 0; v < 8; ++v) {
        const int m = blockM + wm * 32 + mi * 16 + half * 8 + v;
        out[(size_t)m * 768 + n] = (__bf16)(acc[mi][ni][v] + kb);
      }
    }
}

// Kernel 2: W_encI[m, d] = encI_p[m,:768] . bil_w[d,:768]
__global__ void __launch_bounds__(256)
k_wmul(const __bf16* __restrict__ encIp, const float* __restrict__ bilw,
       __bf16* __restrict__ out)
{
  v8f acc[2][4];
  const int blockN = blockIdx.x * BN, blockM = blockIdx.y * BM;
  gemm_tn_tile<__bf16, float>(encIp, 768, bilw, 768, 6400, 768, 768, blockM, blockN, acc);

  const int lane = threadIdx.x & 31, w = threadIdx.x >> 5;
  const int wm = w & 3, wn = w >> 2, half = lane >> 4, lrow = lane & 15;
#pragma unroll
  for (int mi = 0; mi < 2; ++mi)
#pragma unroll
    for (int ni = 0; ni < 4; ++ni) {
      const int n = blockN + wn * 64 + ni * 16 + lrow;
#pragma unroll
      for (int v = 0; v < 8; ++v) {
        const int m = blockM + wm * 32 + mi * 16 + half * 8 + v;
        out[(size_t)m * 768 + n] = (__bf16)acc[mi][ni][v];
      }
    }
}

// Kernel 3 (per batch b): logits[b,t,r] = encT[b,t,:768] . W_encI[b,r,:768] + bil_b + mask[b,t,r]
__global__ void __launch_bounds__(256)
k_bilin(const float* __restrict__ encT, const __bf16* __restrict__ W,
        const float* __restrict__ mask, const float* __restrict__ bilb,
        float* __restrict__ out)
{
  const int b = blockIdx.z;
  v8f acc[2][4];
  gemm_tn_tile<float, __bf16>(encT + (size_t)b * 128 * 768, 768,
                              W + (size_t)b * 100 * 768, 768,
                              128, 100, 768, 0, 0, acc);

  const float bb = bilb[0];
  const int lane = threadIdx.x & 31, w = threadIdx.x >> 5;
  const int wm = w & 3, wn = w >> 2, half = lane >> 4, lrow = lane & 15;
#pragma unroll
  for (int mi = 0; mi < 2; ++mi)
#pragma unroll
    for (int ni = 0; ni < 4; ++ni) {
      const int n = wn * 64 + ni * 16 + lrow;   // r
      if (n < 100) {
#pragma unroll
        for (int v = 0; v < 8; ++v) {
          const int m = wm * 32 + mi * 16 + half * 8 + v;  // t
          const size_t o = (size_t)b * 12800 + (size_t)m * 100 + n;
          out[o] = acc[mi][ni][v] + bb + mask[o];
        }
      }
    }
}

extern "C" void kernel_launch(void* const* d_in, const int* in_sizes, int n_in,
                              void* d_out, int out_size, void* d_ws, size_t ws_size,
                              hipStream_t stream)
{
  const float* encT  = (const float*)d_in[0];  // [64,128,768]
  const float* encI  = (const float*)d_in[1];  // [64,100,2048]
  const float* mask  = (const float*)d_in[2];  // [64,1,128,100]
  const float* K_w   = (const float*)d_in[3];  // [768,2048]
  const float* K_b   = (const float*)d_in[4];  // [768]
  const float* bil_w = (const float*)d_in[5];  // [1,768,768]
  const float* bil_b = (const float*)d_in[6];  // [1]
  float* out = (float*)d_out;                  // [64,128,100]

  __bf16* encI_p = (__bf16*)d_ws;                                       // 6400x768 bf16
  __bf16* WencI  = (__bf16*)((char*)d_ws + (size_t)6400 * 768 * 2);     // 6400x768 bf16

  dim3 blk(256);
  k_proj <<<dim3(768 / BN, 6400 / BM), blk, 0, stream>>>(encI, K_w, K_b, encI_p);
  k_wmul <<<dim3(768 / BN, 6400 / BM), blk, 0, stream>>>(encI_p, bil_w, WencI);
  k_bilin<<<dim3(1, 1, 64),            blk, 0, stream>>>(encT, WencI, mask, bil_b, out);
}
